// VQEmbeddingEMA_7705171329460
// MI455X (gfx1250) — matrix operand order repaired and checked by reading, not compiled, not run on hardware
//
#include <hip/hip_runtime.h>

#define N_BATCH 32
#define T_DIM   1024
#define D_DIM   512
#define M_CODES 2048
#define NT      (N_BATCH * T_DIM)   // 32768
#define EPS_IN  1e-5f
#define LDS_STRIDE 520              // 512 + 8 bf16 pad (keeps 16B align, staggers banks)

typedef __attribute__((ext_vector_type(16))) __bf16 v16bf;
typedef __attribute__((ext_vector_type(8)))  __bf16 v8bf;
typedef __attribute__((ext_vector_type(8)))  float  v8f;
typedef int v4i __attribute__((vector_size(16)));   // matches builtin param pointee

#define AS1 __attribute__((address_space(1)))
#define AS3 __attribute__((address_space(3)))

#if __has_builtin(__builtin_amdgcn_global_load_async_to_lds_b128)
#define HAVE_ASYNC_LDS 1
#endif

__device__ __forceinline__ v16bf combine16(v8bf lo, v8bf hi) {
  v16bf r;
#pragma unroll
  for (int i = 0; i < 8; ++i) { r[i] = lo[i]; r[i + 8] = hi[i]; }
  return r;
}

// ---------------- Kernel 1: instance-norm stats over T per (n,d) ----------------
__global__ __launch_bounds__(256) void k_stats(const float* __restrict__ x,
                                               float* __restrict__ mu,
                                               float* __restrict__ rstd) {
  const int d = blockIdx.x * 256 + threadIdx.x;
  const int n = blockIdx.y;
  const float* xp = x + ((size_t)n * T_DIM) * D_DIM + d;
  float s = 0.f, ss = 0.f;
#pragma unroll 4
  for (int t = 0; t < T_DIM; ++t) {
    float v = xp[(size_t)t * D_DIM];
    s += v; ss += v * v;
  }
  float m   = s * (1.0f / (float)T_DIM);
  float var = (ss - s * m) * (1.0f / (float)(T_DIM - 1));  // unbiased (ddof=1)
  var = fmaxf(var, 0.0f);
  mu[n * D_DIM + d]   = m;
  rstd[n * D_DIM + d] = 1.0f / (sqrtf(var) + EPS_IN);
}

// ---------------- Kernel 2: codebook normalize -> bf16, per-row ||e_n||^2 -------
__global__ __launch_bounds__(256) void k_embprep(const float* __restrict__ emb,
                                                 __bf16* __restrict__ embN,
                                                 float* __restrict__ e2) {
  const int wave = threadIdx.x >> 5, lane = threadIdx.x & 31;
  const int row = blockIdx.x * 8 + wave;
  const float* ep = emb + (size_t)row * D_DIM;
  float v[16]; float ss = 0.f;
#pragma unroll
  for (int i = 0; i < 16; ++i) { v[i] = ep[i * 32 + lane]; ss += v[i] * v[i]; }
#pragma unroll
  for (int m = 16; m >= 1; m >>= 1) ss += __shfl_xor(ss, m, 32);
  const float sc = 1.0f / (sqrtf(ss) + 1e-4f);
  __bf16* op = embN + (size_t)row * D_DIM;
#pragma unroll
  for (int i = 0; i < 16; ++i) op[i * 32 + lane] = (__bf16)(v[i] * sc);
  if (lane == 0) e2[row] = ss * sc * sc;
}

// ---------------- Kernel 3: fused WMMA GEMM (z @ embN^T) + argmin ----------------
__global__ __launch_bounds__(128) void k_argmin(const float* __restrict__ x,
                                                const float* __restrict__ mu,
                                                const float* __restrict__ rstd,
                                                const __bf16* __restrict__ embN,
                                                const float* __restrict__ e2,
                                                int* __restrict__ indices) {
  __shared__ __align__(16) __bf16 zT[64 * LDS_STRIDE];  // 64 z rows   x 512 K
  __shared__ __align__(16) __bf16 eT[64 * LDS_STRIDE];  // 64 codes    x 512 K
  const int tid  = threadIdx.x;
  const int wave = tid >> 5;
  const int lane = tid & 31;
  const int half = lane >> 4;
  const int mrow = lane & 15;
  const int row0 = blockIdx.x * 64;
  const int waveRow = wave * 16;

  // Stage z tile: normalize on the fly, convert to bf16. 8 floats / iter / thread.
  for (int i = tid; i < 64 * 64; i += 128) {
    const int r  = i >> 6;
    const int c8 = (i & 63) << 3;
    const int grow = row0 + r;
    const int n = grow >> 10;                       // T = 1024
    const float4* xp = (const float4*)(x    + (size_t)grow * D_DIM + c8);
    const float4* mp = (const float4*)(mu   + n * D_DIM + c8);
    const float4* sp = (const float4*)(rstd + n * D_DIM + c8);
    float4 x0 = xp[0], x1 = xp[1];
    float4 m0 = mp[0], m1 = mp[1];
    float4 s0 = sp[0], s1 = sp[1];
    union { __bf16 h[8]; uint4 u; } pk;
    pk.h[0] = (__bf16)((x0.x - m0.x) * s0.x);
    pk.h[1] = (__bf16)((x0.y - m0.y) * s0.y);
    pk.h[2] = (__bf16)((x0.z - m0.z) * s0.z);
    pk.h[3] = (__bf16)((x0.w - m0.w) * s0.w);
    pk.h[4] = (__bf16)((x1.x - m1.x) * s1.x);
    pk.h[5] = (__bf16)((x1.y - m1.y) * s1.y);
    pk.h[6] = (__bf16)((x1.z - m1.z) * s1.z);
    pk.h[7] = (__bf16)((x1.w - m1.w) * s1.w);
    *(uint4*)(&zT[r * LDS_STRIDE + c8]) = pk.u;
  }

  float bestD[8]; int bestI[8];
#pragma unroll
  for (int j = 0; j < 8; ++j) { bestD[j] = 3.4e38f; bestI[j] = 0; }

  for (int c = 0; c < M_CODES / 64; ++c) {
    __syncthreads();  // zT ready (iter 0) / previous chunk's compute done
    // Stage 64 codebook rows: CDNA5 async global->LDS path (ASYNCcnt-tracked),
    // bypassing VGPRs; falls back to load+ds_store if builtin is absent.
    for (int i = tid; i < 64 * 64; i += 128) {
      const int r  = i >> 6;
      const int c8 = (i & 63) << 3;
      const __bf16* gsrc = embN + (size_t)(c * 64 + r) * D_DIM + c8;
      __bf16* ldst = &eT[r * LDS_STRIDE + c8];
#ifdef HAVE_ASYNC_LDS
      __builtin_amdgcn_global_load_async_to_lds_b128(
          (AS1 v4i*)(uintptr_t)gsrc,
          (AS3 v4i*)(uintptr_t)ldst,
          0, 0);
#else
      *(uint4*)ldst = *(const uint4*)gsrc;
#endif
    }
#ifdef HAVE_ASYNC_LDS
#if __has_builtin(__builtin_amdgcn_s_wait_asynccnt)
    __builtin_amdgcn_s_wait_asynccnt(0);
#else
    asm volatile("s_wait_asynccnt 0x0" ::: "memory");
#endif
#endif
    __syncthreads();

    v8f acc[4];
#pragma unroll
    for (int s = 0; s < 4; ++s)
#pragma unroll
      for (int j = 0; j < 8; ++j) acc[s][j] = 0.0f;

#pragma unroll 4
    for (int kk = 0; kk < 16; ++kk) {
      // A (16x32 bf16): lanes 0-15 hold K {0..7,16..23}, lanes 16-31 {8..15,24..31}
      const __bf16* ap = &zT[(waveRow + mrow) * LDS_STRIDE + kk * 32 + half * 8];
      v16bf a = combine16(*(const v8bf*)ap, *(const v8bf*)(ap + 16));
#pragma unroll
      for (int s = 0; s < 4; ++s) {
        // B (32x16 bf16): lane = column (code); lanes 0-15 K {0..7,16..23}, 16-31 {8..15,24..31}
        const __bf16* bp = &eT[(s * 16 + mrow) * LDS_STRIDE + kk * 32 + half * 8];
        v16bf b = combine16(*(const v8bf*)bp, *(const v8bf*)(bp + 16));
        acc[s] = __builtin_amdgcn_wmma_f32_16x16x32_bf16(
            false, a, false, b, (short)0, acc[s], false, false);
      }
    }

    // Epilogue: dist = ||e||^2 - 2*dot (per-row ||z||^2 dropped: constant in argmin)
#pragma unroll
    for (int s = 0; s < 4; ++s) {
      const int col = c * 64 + s * 16 + mrow;
      const float ec = e2[col];
#pragma unroll
      for (int j = 0; j < 8; ++j) {
        float d = ec - 2.0f * acc[s][j];
        if (d < bestD[j]) { bestD[j] = d; bestI[j] = col; }  // strict: keeps first min
      }
    }
  }

  // Cross-lane min-reduce within each 16-lane half (row-sharing lanes)
#pragma unroll
  for (int msk = 8; msk >= 1; msk >>= 1) {
#pragma unroll
    for (int j = 0; j < 8; ++j) {
      float od = __shfl_xor(bestD[j], msk, 32);
      int   oi = __shfl_xor(bestI[j], msk, 32);
      if (od < bestD[j] || (od == bestD[j] && oi < bestI[j])) {
        bestD[j] = od; bestI[j] = oi;
      }
    }
  }
  if (mrow == 0) {
#pragma unroll
    for (int j = 0; j < 8; ++j)
      indices[row0 + waveRow + half * 8 + j] = bestI[j];
  }
}

// ---------------- Kernel 4: gather + STE output + loss partials + histogram -----
__global__ __launch_bounds__(256) void k_quant(const float* __restrict__ x,
                                               const float* __restrict__ mu,
                                               const float* __restrict__ rstd,
                                               const float* __restrict__ emb,
                                               const int* __restrict__ indices,
                                               float* __restrict__ out,
                                               unsigned int* __restrict__ counts,
                                               float* __restrict__ partial) {
  __shared__ float red[256];
  const int tid = threadIdx.x;
  const int rowBase = blockIdx.x * 16;
  float lsum = 0.0f;
  for (int rr = 0; rr < 16; ++rr) {
    const int row = rowBase + rr;
    const int idx = indices[row];
    if (tid == 0) atomicAdd(&counts[idx], 1u);
    const int n = row >> 10;
    const float* qp = emb  + (size_t)idx * D_DIM;
    const float* xp = x    + (size_t)row * D_DIM;
    const float* mp = mu   + n * D_DIM;
    const float* sp = rstd + n * D_DIM;
    float* op = out + (size_t)row * D_DIM;
#pragma unroll
    for (int k = 0; k < 2; ++k) {
      const int d = tid + k * 256;
      float q = qp[d];
      float z = (xp[d] - mp[d]) * sp[d];
      float diff = z - q;
      lsum += diff * diff;
      float qs = z + (q - z);          // STE forward, rounding matched to reference
      op[d] = (qs + q) * 0.5f;
    }
  }
  red[tid] = lsum;
  __syncthreads();
  for (int s = 128; s > 0; s >>= 1) {
    if (tid < s) red[tid] += red[tid + s];
    __syncthreads();
  }
  if (tid == 0) partial[blockIdx.x] = red[0];
}

// ---------------- Kernel 5: deterministic final reductions ----------------------
__global__ __launch_bounds__(256) void k_final(const float* __restrict__ partial,
                                               const unsigned int* __restrict__ counts,
                                               float* __restrict__ out_scalars) {
  __shared__ float red[256];
  const int tid = threadIdx.x;
  float ls = 0.f;
  for (int i = tid; i < 2048; i += 256) ls += partial[i];
  red[tid] = ls; __syncthreads();
  for (int s = 128; s > 0; s >>= 1) { if (tid < s) red[tid] += red[tid + s]; __syncthreads(); }
  const float lossSum = red[0];
  __syncthreads();
  float es = 0.f;
  for (int i = tid; i < M_CODES; i += 256) {
    float p = (float)counts[i] * (1.0f / (float)NT);
    es += p * logf(p + 1e-10f);
  }
  red[tid] = es; __syncthreads();
  for (int s = 128; s > 0; s >>= 1) { if (tid < s) red[tid] += red[tid + s]; __syncthreads(); }
  if (tid == 0) {
    out_scalars[0] = lossSum * (1.0f / (float)((size_t)NT * D_DIM));  // commitment loss
    out_scalars[1] = expf(-red[0]);                                   // perplexity
  }
}

extern "C" void kernel_launch(void* const* d_in, const int* in_sizes, int n_in,
                              void* d_out, int out_size, void* d_ws, size_t ws_size,
                              hipStream_t stream) {
  const float* x   = (const float*)d_in[0];   // (32,1024,512) fp32
  const float* emb = (const float*)d_in[1];   // (2048,512)    fp32
  float* out = (float*)d_out;                 // 32*1024*512 + 2 fp32

  char* ws = (char*)d_ws;                     // ~2.4 MB used
  __bf16*       embN    = (__bf16*)(ws + 0);            // 2,097,152 B
  float*        e2      = (float*)(ws + 2097152);       //     8,192 B
  float*        mu      = (float*)(ws + 2105344);       //    65,536 B
  float*        rstd    = (float*)(ws + 2170880);       //    65,536 B
  int*          indices = (int*)(ws + 2236416);         //   131,072 B
  unsigned int* counts  = (unsigned int*)(ws + 2367488);//     8,192 B
  float*        partial = (float*)(ws + 2375680);       //     8,192 B

  (void)hipMemsetAsync(counts, 0, M_CODES * sizeof(unsigned int), stream);
  k_stats  <<<dim3(2, 32), 256, 0, stream>>>(x, mu, rstd);
  k_embprep<<<256,         256, 0, stream>>>(emb, embN, e2);
  k_argmin <<<NT / 64,     128, 0, stream>>>(x, mu, rstd, embN, e2, indices);
  k_quant  <<<NT / 16,     256, 0, stream>>>(x, mu, rstd, emb, indices, out, counts, partial);
  k_final  <<<1,           256, 0, stream>>>(partial, counts, out + (size_t)NT * D_DIM);
}
